// ASMambaBlock_50362786512971
// MI455X (gfx1250) — compile-verified, matrix operand order checked by reading, not run on hardware
//
#include <hip/hip_runtime.h>
#include <hip/hip_bf16.h>
#include <math.h>

typedef _Float16 half_t;
typedef __attribute__((ext_vector_type(16))) _Float16 v16h;
typedef __attribute__((ext_vector_type(8)))  float    v8f;
typedef __attribute__((ext_vector_type(4)))  int      v4i;

#define DD     384
#define NN_TOK 1024
#define BB     8
#define ROWS   (BB * NN_TOK)   // 8192
#define DIN    768
#define DSTATE 64
#define DTRANK 24
#define XDBL   152             // dt_rank + 2*d_state

// ---------------------------------------------------------------------------
// CDNA5 async global->LDS staging (ASYNCcnt path), with sync fallback.
// Builtin signature (from clang diagnostic): (AS1 v4i*, AS3 v4i*, imm, imm).
// ---------------------------------------------------------------------------
#if __has_builtin(__builtin_amdgcn_global_load_async_to_lds_b128)
#define HAVE_ASYNC_LDS 1
#else
#define HAVE_ASYNC_LDS 0
#endif

typedef __attribute__((address_space(1))) v4i* gv4i_ptr;
typedef __attribute__((address_space(3))) v4i* lv4i_ptr;

__device__ __forceinline__ void stage_b128(const half_t* g, half_t* l) {
#if HAVE_ASYNC_LDS
  __builtin_amdgcn_global_load_async_to_lds_b128((gv4i_ptr)g, (lv4i_ptr)l, 0, 0);
#else
  *(uint4*)l = *(const uint4*)g;
#endif
}

__device__ __forceinline__ void wait_async_le3() {
#if HAVE_ASYNC_LDS
#if __has_builtin(__builtin_amdgcn_s_wait_asynccnt)
  __builtin_amdgcn_s_wait_asynccnt(3);
#else
  asm volatile("s_wait_asynccnt 3" ::: "memory");
#endif
#endif
}

__device__ __forceinline__ void wait_async_le0() {
#if HAVE_ASYNC_LDS
#if __has_builtin(__builtin_amdgcn_s_wait_asynccnt)
  __builtin_amdgcn_s_wait_asynccnt(0);
#else
  asm volatile("s_wait_asynccnt 0" ::: "memory");
#endif
#endif
}

// ---------------------------------------------------------------------------
// Scan-direction permutations (side = sqrt(1024) = 32)
// ---------------------------------------------------------------------------
__device__ __forceinline__ int perm_idx(int i, int mode) {
  switch (mode) {
    case 1:  return (i & 31) * 32 + (i >> 5);
    case 2:  return 1023 - i;
    case 3:  return 1023 - ((i & 31) * 32 + (i >> 5));
    default: return i;
  }
}

// ---------------------------------------------------------------------------
// Weight pre-convert: src f32 [Nn,K] -> dst f16 [Np,Kp], zero padded.
// Guarantees all GEMM staging is unconditional, aligned, in-bounds, and that
// K-pad columns contribute exact zeros.
// ---------------------------------------------------------------------------
__global__ void cvt_w_kernel(const float* __restrict__ src, half_t* __restrict__ dst,
                             int Nn, int K, int Np, int Kp) {
  const int i = blockIdx.x * blockDim.x + threadIdx.x;
  if (i >= Np * Kp) return;
  const int n = i / Kp, k = i % Kp;
  dst[i] = (n < Nn && k < K) ? (half_t)src[(long)n * K + k] : (half_t)0.f;
}

// ---------------------------------------------------------------------------
// WMMA GEMM: C[M,Nn] = A[M,lda](f16, K cols, K%32==0) @ Wh[Np,K](f16,padded)^T
// Block tile 64(M) x 128(N), 8 waves as 2x4, wave tile 32x32 (2x2 WMMA accs).
// Double-buffered LDS tiles, async b128 global->LDS staging, software pipeline:
//   issue(k+1) -> s_wait_asynccnt 3 (chunk k done, in-order) -> barrier ->
//   wmma(k) -> barrier (protects buffer reuse).
// epi: 0 = (+bias) store f32 (+optional f16 mirror of first n16 cols)
//      2 = softplus(acc + bias) store f32
//      3 = C += wrouter[(row>>10)*4 + dir] * acc
// perm_mode >= 0: gather A rows through scan-direction permutation.
// ---------------------------------------------------------------------------
__launch_bounds__(256)
__global__ void gemm_wmma_kernel(const half_t* __restrict__ A, int lda,
                                 const half_t* __restrict__ Wh,
                                 const float* __restrict__ bias,
                                 float* __restrict__ C, int ldc,
                                 half_t* __restrict__ C16, int ldc16, int n16,
                                 int M, int Nn, int K,
                                 int epi, const float* __restrict__ wrouter, int dir,
                                 int perm_mode) {
  __shared__ __align__(16) half_t As[2][64][40];
  __shared__ __align__(16) half_t Bs[2][128][40];

  const int tid    = threadIdx.x;
  const int lane   = tid & 31;
  const int wid    = tid >> 5;
  const int waveM  = wid & 1;    // 0..1
  const int waveN  = wid >> 1;   // 0..3
  const int half16 = lane >> 4;  // 0/1
  const int l15    = lane & 15;
  const int m0     = blockIdx.x * 64;
  const int n0     = blockIdx.y * 128;

  v8f acc[2][2];
  acc[0][0] = {}; acc[0][1] = {}; acc[1][0] = {}; acc[1][1] = {};

  // A staging: 64 rows x 32 halves; one b128 per thread.
  const int am = tid >> 2;
  const int ak = (tid & 3) * 8;
  int arow = m0 + am;
  if (perm_mode >= 0) {
    int i = arow & (NN_TOK - 1);
    arow = (arow & ~(NN_TOK - 1)) + perm_idx(i, perm_mode);
  }
  const half_t* aptr = A + (long)arow * lda + ak;

  // W staging: 128 n-rows x 32 halves; two b128 per thread.
  const int bn = tid >> 1;
  const int bk = (tid & 1) * 16;
  const half_t* wptr = Wh + (long)(n0 + bn) * K + bk;

  const int nK = K >> 5;  // K % 32 == 0 by construction

  // prologue: stage chunk 0 into buffer 0
  stage_b128(aptr, &As[0][am][ak]);
  stage_b128(wptr,     &Bs[0][bn][bk]);
  stage_b128(wptr + 8, &Bs[0][bn][bk + 8]);

  for (int kc = 0; kc < nK; ++kc) {
    const int cur = kc & 1;
    if (kc + 1 < nK) {  // stage next chunk into the other buffer
      const int kb = (kc + 1) << 5;
      stage_b128(aptr + kb, &As[cur ^ 1][am][ak]);
      stage_b128(wptr + kb,     &Bs[cur ^ 1][bn][bk]);
      stage_b128(wptr + kb + 8, &Bs[cur ^ 1][bn][bk + 8]);
      wait_async_le3();   // chunk kc's 3 loads complete (in-order per wave)
    } else {
      wait_async_le0();
    }
    __syncthreads();

    // Pack fragments per CDNA5 16-bit A/B layout: lane m/n = lane&15,
    // K indices split across lane halves, 2 consecutive K per 32-bit reg.
    union Frag { v16h v; unsigned u[8]; };
    Frag af[2], bf[2];
#pragma unroll
    for (int t = 0; t < 2; ++t) {
      const int mrow = waveM * 32 + t * 16 + l15;
      const int nrow = waveN * 32 + t * 16 + l15;
#pragma unroll
      for (int v = 0; v < 8; ++v) {
        const int kk = (v < 4 ? 2 * v : 16 + 2 * (v - 4)) + half16 * 8;
        af[t].u[v] = *(const unsigned*)&As[cur][mrow][kk];
        bf[t].u[v] = *(const unsigned*)&Bs[cur][nrow][kk];
      }
    }
#pragma unroll
    for (int mt = 0; mt < 2; ++mt)
#pragma unroll
      for (int nt = 0; nt < 2; ++nt)
        acc[mt][nt] = __builtin_amdgcn_wmma_f32_16x16x32_f16(
            false, af[mt].v, false, bf[nt].v, (short)0, acc[mt][nt], false, false);

    __syncthreads();  // all waves done reading buf[cur] before it is restaged
  }

  // Epilogue. D layout: elem v -> row = v + half16*8 within the 16x16 tile.
#pragma unroll
  for (int mt = 0; mt < 2; ++mt) {
#pragma unroll
    for (int nt = 0; nt < 2; ++nt) {
      const int col = n0 + waveN * 32 + nt * 16 + l15;
      if (col >= Nn) continue;
#pragma unroll
      for (int v = 0; v < 8; ++v) {
        const int row = m0 + waveM * 32 + mt * 16 + v + half16 * 8;
        if (row >= M) continue;
        float val = acc[mt][nt][v];
        if (epi == 2) {
          if (bias) val += bias[col];
          val = (val > 20.f) ? val : log1pf(__expf(val));
          C[(long)row * ldc + col] = val;
        } else if (epi == 3) {
          const float s = wrouter[((row >> 10) << 2) + dir];
          C[(long)row * ldc + col] += s * val;
        } else {
          if (bias) val += bias[col];
          C[(long)row * ldc + col] = val;
          if (C16 && col < n16) C16[(long)row * ldc16 + col] = (half_t)val;
        }
      }
    }
  }
}

// ---------------------------------------------------------------------------
// Router: mean pool -> GELU MLP -> softmax over 4 directions. One block/batch.
// ---------------------------------------------------------------------------
__global__ void router_kernel(const float* __restrict__ x,
                              const float* __restrict__ w1, const float* __restrict__ b1,
                              const float* __restrict__ w2, const float* __restrict__ b2,
                              float* __restrict__ wout) {
  const int b = blockIdx.x, tid = threadIdx.x;
  __shared__ float g[DD];
  __shared__ float hbuf[96];
  __shared__ float sc[4];
  for (int d = tid; d < DD; d += blockDim.x) {
    float s = 0.f;
    for (int n = 0; n < NN_TOK; ++n) s += x[((long)b * NN_TOK + n) * DD + d];
    g[d] = s * (1.f / NN_TOK);
  }
  __syncthreads();
  for (int j = tid; j < 96; j += blockDim.x) {
    float s = b1[j];
    for (int d = 0; d < DD; ++d) s += g[d] * w1[j * DD + d];
    hbuf[j] = 0.5f * s * (1.f + erff(s * 0.70710678118f));  // exact GELU
  }
  __syncthreads();
  if (tid < 4) {
    float s = b2[tid];
    for (int j = 0; j < 96; ++j) s += hbuf[j] * w2[tid * 96 + j];
    sc[tid] = s;
  }
  __syncthreads();
  if (tid == 0) {
    float m = fmaxf(fmaxf(sc[0], sc[1]), fmaxf(sc[2], sc[3]));
    float e[4], sum = 0.f;
    for (int k = 0; k < 4; ++k) { e[k] = __expf(sc[k] - m); sum += e[k]; }
    const float inv = 1.f / sum;
    for (int k = 0; k < 4; ++k) wout[b * 4 + k] = e[k] * inv;
  }
}

// ---------------------------------------------------------------------------
// LayerNorm: row = in0 + gate*in1 (optional), one wave per 384-elem row,
// shuffle reductions; optional f32 and f16 outputs.
// ---------------------------------------------------------------------------
__launch_bounds__(256)
__global__ void ln_kernel(const float* __restrict__ in0,
                          const float* __restrict__ in1,
                          const float* __restrict__ gate,
                          const float* __restrict__ w, const float* __restrict__ b,
                          float* __restrict__ out32, half_t* __restrict__ out16,
                          int rows) {
  const int row = blockIdx.x * 8 + (threadIdx.x >> 5);
  if (row >= rows) return;
  const int lane = threadIdx.x & 31;
  const float g = gate ? *gate : 1.f;
  float vals[12];
  float sum = 0.f, sq = 0.f;
#pragma unroll
  for (int k = 0; k < 12; ++k) {
    const int j = lane + k * 32;
    float v = in0[(long)row * DD + j];
    if (in1) v += g * in1[(long)row * DD + j];
    vals[k] = v; sum += v; sq += v * v;
  }
#pragma unroll
  for (int m = 16; m >= 1; m >>= 1) { sum += __shfl_xor(sum, m); sq += __shfl_xor(sq, m); }
  const float mu  = sum * (1.f / DD);
  const float var = sq * (1.f / DD) - mu * mu;
  const float rs  = rsqrtf(var + 1e-5f);
#pragma unroll
  for (int k = 0; k < 12; ++k) {
    const int j = lane + k * 32;
    const float o = (vals[k] - mu) * rs * w[j] + b[j];
    if (out32) out32[(long)row * DD + j] = o;
    if (out16) out16[(long)row * DD + j] = (half_t)o;
  }
}

// ---------------------------------------------------------------------------
// Depthwise causal conv (k=4) + SiLU over xin = xz[:, :768]
// ---------------------------------------------------------------------------
__global__ void conv_silu_kernel(const float* __restrict__ xz,
                                 const float* __restrict__ cw,
                                 const float* __restrict__ cb,
                                 float* __restrict__ u, half_t* __restrict__ uh) {
  const int idx = blockIdx.x * blockDim.x + threadIdx.x;
  if (idx >= ROWS * DIN) return;
  const int c = idx % DIN;
  const int t = idx / DIN;          // row = b*1024 + i
  const int i = t & (NN_TOK - 1);
  float acc = cb[c];
#pragma unroll
  for (int j = 0; j < 4; ++j) {
    const int ti = i - 3 + j;
    if (ti >= 0) acc += cw[c * 4 + j] * xz[(long)(t - 3 + j) * 1536 + c];
  }
  const float s = acc / (1.f + __expf(-acc));
  u[idx]  = s;
  uh[idx] = (half_t)s;
}

// ---------------------------------------------------------------------------
// Selective scan. Grid (6 channel-blocks, 8 batches), 256 threads.
// 2 lanes per channel, 32 states each held in VGPRs; B/C/dt/u staged via LDS.
// Fused: y = scan + u*D_skip; out = y * silu(z) -> f16 for out_proj GEMM.
// ---------------------------------------------------------------------------
__launch_bounds__(256)
__global__ void scan_kernel(const float* __restrict__ dt,
                            const float* __restrict__ u,
                            const float* __restrict__ xdbl,
                            const float* __restrict__ A_log,
                            const float* __restrict__ Dskip,
                            const float* __restrict__ xz,
                            half_t* __restrict__ yz) {
  const int tid = threadIdx.x;
  const int cb  = blockIdx.x * 128;
  const int b   = blockIdx.y;
  const int cl  = tid >> 1;   // channel within block
  const int sh  = tid & 1;    // state half
  const int c   = cb + cl;

  float a[32], h[32];
#pragma unroll
  for (int s = 0; s < 32; ++s) {
    a[s] = -__expf(A_log[c * DSTATE + sh * 32 + s]);
    h[s] = 0.f;
  }
  const float dsk = Dskip[c];

  __shared__ float sB[64], sC[64], sdt[128], su[128];
  for (int i = 0; i < NN_TOK; ++i) {
    const int row = (b << 10) + i;
    if (tid < 128) {
      sdt[tid] = dt[(long)row * DIN + cb + tid];
      su[tid]  = u[(long)row * DIN + cb + tid];
    } else if (tid < 192) {
      sB[tid - 128] = xdbl[(long)row * XDBL + DTRANK + (tid - 128)];
    } else {
      sC[tid - 192] = xdbl[(long)row * XDBL + DTRANK + DSTATE + (tid - 192)];
    }
    __syncthreads();
    const float dtc = sdt[cl], uc = su[cl];
    const float dtu = dtc * uc;
    float accv = 0.f;
#pragma unroll
    for (int s = 0; s < 32; ++s) {
      const float dA = __expf(dtc * a[s]);
      h[s] = dA * h[s] + dtu * sB[sh * 32 + s];
      accv += h[s] * sC[sh * 32 + s];
    }
    accv += __shfl_xor(accv, 1);  // combine the two state halves
    if (sh == 0) {
      const float y  = accv + uc * dsk;
      const float z  = xz[(long)row * 1536 + DIN + c];
      const float sz = z / (1.f + __expf(-z));
      yz[(long)row * DIN + c] = (half_t)(y * sz);
    }
    __syncthreads();
  }
}

// ---------------------------------------------------------------------------
// Windowed attention core: 4-token windows, 4 heads x 96 dims.
// One wave per (window, head); lane covers 3 dims; full-wave xor reductions.
// ---------------------------------------------------------------------------
__launch_bounds__(256)
__global__ void attn_kernel(const float* __restrict__ qkv, half_t* __restrict__ o16) {
  const int task = blockIdx.x * 8 + (threadIdx.x >> 5);  // 0..8191
  const int lane = threadIdx.x & 31;
  const int head = task & 3;
  const int win  = task >> 2;         // b*256 + window
  const int tok0 = win * 4;           // = b*1024 + wloc*4
  const int dofs = head * 96 + lane * 3;

  float q[4][3], k[4][3], v[4][3];
#pragma unroll
  for (int t = 0; t < 4; ++t) {
    const long base = (long)(tok0 + t) * 1152;
#pragma unroll
    for (int j = 0; j < 3; ++j) {
      q[t][j] = qkv[base + dofs + j];
      k[t][j] = qkv[base + 384 + dofs + j];
      v[t][j] = qkv[base + 768 + dofs + j];
    }
  }
  float att[4][4];
  const float scale = 0.1020620726159658f;  // 1/sqrt(96)
#pragma unroll
  for (int qi = 0; qi < 4; ++qi)
#pragma unroll
    for (int ki = 0; ki < 4; ++ki) {
      float p = q[qi][0] * k[ki][0] + q[qi][1] * k[ki][1] + q[qi][2] * k[ki][2];
#pragma unroll
      for (int m = 16; m >= 1; m >>= 1) p += __shfl_xor(p, m);
      att[qi][ki] = p * scale;
    }
#pragma unroll
  for (int qi = 0; qi < 4; ++qi) {
    float m = fmaxf(fmaxf(att[qi][0], att[qi][1]), fmaxf(att[qi][2], att[qi][3]));
    float e[4], s = 0.f;
#pragma unroll
    for (int ki = 0; ki < 4; ++ki) { e[ki] = __expf(att[qi][ki] - m); s += e[ki]; }
    const float inv = 1.f / s;
#pragma unroll
    for (int ki = 0; ki < 4; ++ki) att[qi][ki] = e[ki] * inv;
  }
#pragma unroll
  for (int qi = 0; qi < 4; ++qi)
#pragma unroll
    for (int j = 0; j < 3; ++j) {
      const float o = att[qi][0] * v[0][j] + att[qi][1] * v[1][j] +
                      att[qi][2] * v[2][j] + att[qi][3] * v[3][j];
      o16[(long)(tok0 + qi) * DD + dofs + j] = (half_t)o;
    }
}

// ---------------------------------------------------------------------------
extern "C" void kernel_launch(void* const* d_in, const int* in_sizes, int n_in,
                              void* d_out, int out_size, void* d_ws, size_t ws_size,
                              hipStream_t stream) {
  const float* x         = (const float*)d_in[0];
  const float* r_w1      = (const float*)d_in[1];
  const float* r_b1      = (const float*)d_in[2];
  const float* r_w2      = (const float*)d_in[3];
  const float* r_b2      = (const float*)d_in[4];
  const float* ln1_w     = (const float*)d_in[5];
  const float* ln1_b     = (const float*)d_in[6];
  const float* ln2_w     = (const float*)d_in[7];
  const float* ln2_b     = (const float*)d_in[8];
  const float* in_proj_w = (const float*)d_in[9];
  const float* conv_w    = (const float*)d_in[10];
  const float* conv_b    = (const float*)d_in[11];
  const float* x_proj_w  = (const float*)d_in[12];
  const float* dt_proj_w = (const float*)d_in[13];
  const float* dt_proj_b = (const float*)d_in[14];
  const float* A_log     = (const float*)d_in[15];
  const float* D_skip    = (const float*)d_in[16];
  const float* out_proj_w= (const float*)d_in[17];
  const float* qkv_w     = (const float*)d_in[18];
  const float* qkv_b     = (const float*)d_in[19];
  const float* ao_w      = (const float*)d_in[20];
  const float* ao_b      = (const float*)d_in[21];
  const float* gate      = (const float*)d_in[22];
  const float* lng_w     = (const float*)d_in[23];
  const float* lng_b     = (const float*)d_in[24];

  char* ws = (char*)d_ws;
  size_t off = 0;
  auto take = [&](size_t bytes) -> char* {
    char* p = ws + off;
    off += (bytes + 255) & ~(size_t)255;
    return p;
  };
  // activations
  float*  xz    = (float*)take((size_t)ROWS * 1536 * 4);  // reused: qkv
  float*  ubuf  = (float*)take((size_t)ROWS * DIN * 4);   // reused: ao
  half_t* uh    = (half_t*)take((size_t)ROWS * DIN * 2);  // reused: attn o16
  float*  dtb   = (float*)take((size_t)ROWS * DIN * 4);   // reused: xn2 f32
  float*  xdbl  = (float*)take((size_t)ROWS * XDBL * 4);
  half_t* xdblh = (half_t*)take((size_t)ROWS * 32 * 2);   // lda=32, cols 24..31 benign
  half_t* yzb   = (half_t*)take((size_t)ROWS * DIN * 2);  // reused: xn2 f16
  float*  fused = (float*)take((size_t)ROWS * DD * 4);
  half_t* xnh   = (half_t*)take((size_t)ROWS * DD * 2);
  float*  wrt   = (float*)take(64 * 4);
  // f16 weights, zero-padded to [Np, Kp] (Kp % 32 == 0, Np covers N tiles)
  half_t* w_in  = (half_t*)take((size_t)1536 * 384 * 2);
  half_t* w_xp  = (half_t*)take((size_t)256  * 768 * 2);  // N 152 -> 256
  half_t* w_dt  = (half_t*)take((size_t)768  * 32  * 2);  // K 24 -> 32
  half_t* w_out = (half_t*)take((size_t)384  * 768 * 2);
  half_t* w_qkv = (half_t*)take((size_t)1152 * 384 * 2);
  half_t* w_ao  = (half_t*)take((size_t)384  * 384 * 2);
  // stage-2 aliases (original users are dead by then)
  float*  qkvb = xz;
  float*  xn2f = dtb;
  half_t* xn2h = yzb;
  half_t* o16b = uh;
  float*  aob  = ubuf;

  // 0) weight conversion (f32 -> padded f16)
  auto cvt = [&](const float* s, half_t* dptr, int Nn, int K, int Np, int Kp) {
    const int n = Np * Kp;
    cvt_w_kernel<<<(n + 255) / 256, 256, 0, stream>>>(s, dptr, Nn, K, Np, Kp);
  };
  cvt(in_proj_w,  w_in,  1536, 384, 1536, 384);
  cvt(x_proj_w,   w_xp,  152,  768, 256,  768);
  cvt(dt_proj_w,  w_dt,  768,  24,  768,  32);
  cvt(out_proj_w, w_out, 384,  768, 384,  768);
  cvt(qkv_w,      w_qkv, 1152, 384, 1152, 384);
  cvt(ao_w,       w_ao,  384,  384, 384,  384);

  // 1) router weights
  router_kernel<<<BB, 128, 0, stream>>>(x, r_w1, r_b1, r_w2, r_b2, wrt);
  // 2) LN1 -> f16 activations for the mamba GEMMs
  ln_kernel<<<ROWS / 8, 256, 0, stream>>>(x, nullptr, nullptr, ln1_w, ln1_b,
                                          nullptr, xnh, ROWS);
  // 3) zero the direction accumulator
  (void)hipMemsetAsync(fused, 0, (size_t)ROWS * DD * 4, stream);

  for (int d = 0; d < 4; ++d) {
    // in_proj: [8192,1536] = perm(xn) @ W^T
    gemm_wmma_kernel<<<dim3(ROWS / 64, 1536 / 128), 256, 0, stream>>>(
        xnh, DD, w_in, nullptr, xz, 1536, nullptr, 0, 0,
        ROWS, 1536, DD, 0, nullptr, 0, d);
    // depthwise conv + silu
    conv_silu_kernel<<<(ROWS * DIN) / 256, 256, 0, stream>>>(xz, conv_w, conv_b, ubuf, uh);
    // x_proj: [8192,152] + f16 mirror of cols 0..31 (dt-rank slice + benign pad)
    gemm_wmma_kernel<<<dim3(ROWS / 64, 2), 256, 0, stream>>>(
        uh, DIN, w_xp, nullptr, xdbl, XDBL, xdblh, 32, 32,
        ROWS, XDBL, DIN, 0, nullptr, 0, -1);
    // dt: softplus([8192,32] @ W^T + b) -> [8192,768]; pad cols hit zero weights
    gemm_wmma_kernel<<<dim3(ROWS / 64, DIN / 128), 256, 0, stream>>>(
        xdblh, 32, w_dt, dt_proj_b, dtb, DIN, nullptr, 0, 0,
        ROWS, DIN, 32, 2, nullptr, 0, -1);
    // selective scan + D-skip + silu(z) gate -> yz f16
    scan_kernel<<<dim3(DIN / 128, BB), 256, 0, stream>>>(
        dtb, ubuf, xdbl, A_log, D_skip, xz, yzb);
    // out_proj with router-weighted accumulate into fused
    gemm_wmma_kernel<<<dim3(ROWS / 64, DD / 128), 256, 0, stream>>>(
        yzb, DIN, w_out, nullptr, fused, DD, nullptr, 0, 0,
        ROWS, DD, DIN, 3, wrt, d, -1);
  }

  // 4) residual + LN2 (f32 + f16)
  ln_kernel<<<ROWS / 8, 256, 0, stream>>>(x, fused, nullptr, ln2_w, ln2_b,
                                          xn2f, xn2h, ROWS);
  // 5) QKV projection (+bias)
  gemm_wmma_kernel<<<dim3(ROWS / 64, 1152 / 128), 256, 0, stream>>>(
      xn2h, DD, w_qkv, qkv_b, qkvb, 1152, nullptr, 0, 0,
      ROWS, 1152, DD, 0, nullptr, 0, -1);
  // 6) windowed attention core -> o f16
  attn_kernel<<<ROWS / 8, 256, 0, stream>>>(qkvb, o16b);
  // 7) attention output projection (+bias)
  gemm_wmma_kernel<<<dim3(ROWS / 64, DD / 128), 256, 0, stream>>>(
      o16b, DD, w_ao, ao_b, aob, DD, nullptr, 0, 0,
      ROWS, DD, DD, 0, nullptr, 0, -1);
  // 8) final LN of xn2 + gate*ao -> d_out
  ln_kernel<<<ROWS / 8, 256, 0, stream>>>(xn2f, aob, gate, lng_w, lng_b,
                                          (float*)d_out, nullptr, ROWS);
}
